// MasterBinaryMemoryRNNCell_56873956934279
// MI455X (gfx1250) — matrix operand order repaired and checked by reading, not compiled, not run on hardware
//
#include <hip/hip_runtime.h>
#include <hip/hip_bf16.h>
#include <math.h>

#define B_DIM 4096
#define H_DIM 2048
#define I_DIM 2048
#define S_DIM 3
#define LN_EPS 1e-5f

#define BM 128
#define BN 64
#define BK 32
#define LDA (BK + 2)   // half stride (even -> k-pairs stay 4B aligned)

typedef __attribute__((ext_vector_type(16))) _Float16 v16h;
typedef __attribute__((ext_vector_type(8)))  float    v8f;

union Frag16 { v16h v; unsigned int u[8]; _Float16 h[16]; };

// ---------------------------------------------------------------------------
// Kernel 1: gate logits + gumbel softmax + straight-through hard gate
// ---------------------------------------------------------------------------
__global__ __launch_bounds__(256)
void gate_kernel(const float* __restrict__ prev_master_h,
                 const float* __restrict__ gate_W,   // [S,H]
                 const float* __restrict__ gate_b,   // [S]
                 const float* __restrict__ gumbel,   // [B,S]
                 float* __restrict__ gate_out)       // [B,S]
{
    __shared__ float red[3][256];
    const int b   = blockIdx.x;
    const int tid = threadIdx.x;

    float p0 = 0.f, p1 = 0.f, p2 = 0.f;
    for (int h = tid; h < H_DIM; h += 256) {
        const float v = prev_master_h[(size_t)b * H_DIM + h];
        p0 += v * gate_W[0 * H_DIM + h];
        p1 += v * gate_W[1 * H_DIM + h];
        p2 += v * gate_W[2 * H_DIM + h];
    }
    red[0][tid] = p0; red[1][tid] = p1; red[2][tid] = p2;
    __syncthreads();
    for (int off = 128; off > 0; off >>= 1) {
        if (tid < off) {
            red[0][tid] += red[0][tid + off];
            red[1][tid] += red[1][tid + off];
            red[2][tid] += red[2][tid + off];
        }
        __syncthreads();
    }
    if (tid == 0) {
        float z[3], mx = -1e30f;
        #pragma unroll
        for (int s = 0; s < 3; ++s) {
            z[s] = red[s][0] + gate_b[s] + gumbel[b * 3 + s];  // TAU == 1
            mx = fmaxf(mx, z[s]);
        }
        float e[3], sum = 0.f;
        #pragma unroll
        for (int s = 0; s < 3; ++s) { e[s] = expf(z[s] - mx); sum += e[s]; }
        float ys[3]; int arg = 0; float best = -1.f;
        #pragma unroll
        for (int s = 0; s < 3; ++s) {
            ys[s] = e[s] / sum;
            if (ys[s] > best) { best = ys[s]; arg = s; }   // first max, like argmax
        }
        #pragma unroll
        for (int s = 0; s < 3; ++s) {
            const float hard = (s == arg) ? 1.f : 0.f;
            gate_out[b * 3 + s] = (hard + ys[s]) - ys[s];  // straight-through value
        }
    }
}

// ---------------------------------------------------------------------------
// Kernel 2: new_sub[s] = tanh(x W_ih[s]^T + prev_sub_h[s] W_hh[s]^T + biases)
// Block tile 128x64, 8 waves of 32x32 (2x2 WMMA), two K phases.
// ---------------------------------------------------------------------------
__global__ __launch_bounds__(256)
void sub_gemm_kernel(const float* __restrict__ x,          // [B,I]
                     const float* __restrict__ prev_sub_h, // [S,B,H]
                     const float* __restrict__ W_ih,       // [S,H,I]
                     const float* __restrict__ W_hh,       // [S,H,H]
                     const float* __restrict__ b_ih,       // [S,H]
                     const float* __restrict__ b_hh,       // [S,H]
                     float* __restrict__ new_sub)          // [S,B,H]
{
    __shared__ _Float16 As[BM][LDA];
    __shared__ _Float16 Bs[BN][LDA];

    const int tid  = threadIdx.x;
    const int lane = tid & 31;
    const int wid  = tid >> 5;
    const int wm   = (wid & 3) * 32;   // wave M offset in block tile
    const int wn   = (wid >> 2) * 32;  // wave N offset in block tile
    const int s    = blockIdx.z;
    const int m0   = blockIdx.x * BM;
    const int n0   = blockIdx.y * BN;

    const v8f vzero = {0.f,0.f,0.f,0.f,0.f,0.f,0.f,0.f};
    v8f acc[2][2];
    #pragma unroll
    for (int i = 0; i < 2; ++i)
        #pragma unroll
        for (int j = 0; j < 2; ++j) acc[i][j] = vzero;

    const int mrow  = lane & 15;
    const int half8 = (lane >> 4) * 8;   // A k-group offset
    const int kb    = (lane >> 4) * 16;  // unused; B loads use half8 like A

    for (int phase = 0; phase < 2; ++phase) {
        const float* Ag = (phase == 0) ? x
                                       : prev_sub_h + (size_t)s * B_DIM * H_DIM;
        const float* Bg = ((phase == 0) ? W_ih : W_hh) + (size_t)s * H_DIM * I_DIM;
        const int K = (phase == 0) ? I_DIM : H_DIM;  // both 2048

        for (int k0 = 0; k0 < K; k0 += BK) {
            __syncthreads();
            // stage A tile (128x32 f32 -> f16)
            #pragma unroll
            for (int r = 0; r < 4; ++r) {
                const int lin = tid * 4 + r * 1024;
                const int m = lin >> 5, kk = lin & 31;
                const float4 f = *(const float4*)(Ag + (size_t)(m0 + m) * K + k0 + kk);
                As[m][kk + 0] = (_Float16)f.x; As[m][kk + 1] = (_Float16)f.y;
                As[m][kk + 2] = (_Float16)f.z; As[m][kk + 3] = (_Float16)f.w;
            }
            // stage B tile (64x32 f32 -> f16), rows are W[n, k] (contiguous in k)
            #pragma unroll
            for (int r = 0; r < 2; ++r) {
                const int lin = tid * 4 + r * 1024;
                const int n = lin >> 5, kk = lin & 31;
                const float4 f = *(const float4*)(Bg + (size_t)(n0 + n) * K + k0 + kk);
                Bs[n][kk + 0] = (_Float16)f.x; Bs[n][kk + 1] = (_Float16)f.y;
                Bs[n][kk + 2] = (_Float16)f.z; Bs[n][kk + 3] = (_Float16)f.w;
            }
            __syncthreads();

            Frag16 a[2], b[2];
            #pragma unroll
            for (int mt = 0; mt < 2; ++mt) {
                const int row = wm + mt * 16 + mrow;
                #pragma unroll
                for (int j = 0; j < 8; ++j) {
                    const int k = ((j < 4) ? 2 * j : 16 + 2 * (j - 4)) + half8;
                    a[mt].u[j] = *(const unsigned int*)&As[row][k];
                }
            }
            #pragma unroll
            for (int nt = 0; nt < 2; ++nt) {
                const int col = wn + nt * 16 + mrow;
                #pragma unroll
                for (int j = 0; j < 8; ++j)
                    b[nt].u[j] = *(const unsigned int*)&Bs[col][((j < 4) ? 2 * j : 16 + 2 * (j - 4)) + half8];
            }
            #pragma unroll
            for (int mt = 0; mt < 2; ++mt)
                #pragma unroll
                for (int nt = 0; nt < 2; ++nt)
                    acc[mt][nt] = __builtin_amdgcn_wmma_f32_16x16x32_f16(
                        false, a[mt].v, false, b[nt].v,
                        (short)0, acc[mt][nt], false, false);
        }
    }

    // epilogue: bias + tanh
    const int mhi = (lane >> 4) * 8;
    #pragma unroll
    for (int mt = 0; mt < 2; ++mt) {
        #pragma unroll
        for (int nt = 0; nt < 2; ++nt) {
            const int n = n0 + wn + nt * 16 + mrow;
            const float bias = b_ih[s * H_DIM + n] + b_hh[s * H_DIM + n];
            #pragma unroll
            for (int j = 0; j < 8; ++j) {
                const int m = m0 + wm + mt * 16 + j + mhi;
                new_sub[(size_t)s * B_DIM * H_DIM + (size_t)m * H_DIM + n] =
                    tanhf(acc[mt][nt][j] + bias);
            }
        }
    }
}

// ---------------------------------------------------------------------------
// Kernel 3: pre = prev_master_h + (gate-weighted new_sub) @ fusion_W^T + fusion_b
// A tile (weighted) materialized during LDS staging. Writes "pre" into the
// master-output slice (in-place scratch; kernel 4 overwrites it with sigmoid).
// ---------------------------------------------------------------------------
__global__ __launch_bounds__(256)
void fusion_gemm_kernel(const float* __restrict__ new_sub,      // [S,B,H]
                        const float* __restrict__ gate,         // [B,S]
                        const float* __restrict__ fusion_W,     // [H,H]
                        const float* __restrict__ fusion_b,     // [H]
                        const float* __restrict__ prev_master_h,// [B,H]
                        float* __restrict__ pre)                // [B,H]
{
    __shared__ _Float16 As[BM][LDA];
    __shared__ _Float16 Bs[BN][LDA];

    const int tid  = threadIdx.x;
    const int lane = tid & 31;
    const int wid  = tid >> 5;
    const int wm   = (wid & 3) * 32;
    const int wn   = (wid >> 2) * 32;
    const int m0   = blockIdx.x * BM;
    const int n0   = blockIdx.y * BN;

    const v8f vzero = {0.f,0.f,0.f,0.f,0.f,0.f,0.f,0.f};
    v8f acc[2][2];
    #pragma unroll
    for (int i = 0; i < 2; ++i)
        #pragma unroll
        for (int j = 0; j < 2; ++j) acc[i][j] = vzero;

    const int mrow  = lane & 15;
    const int half8 = (lane >> 4) * 8;
    const int kb    = (lane >> 4) * 16;
    const size_t SBH = (size_t)B_DIM * H_DIM;

    for (int k0 = 0; k0 < H_DIM; k0 += BK) {
        __syncthreads();
        // stage A = sum_s gate[b,s] * new_sub[s,b,k], converted to f16
        #pragma unroll
        for (int r = 0; r < 4; ++r) {
            const int lin = tid * 4 + r * 1024;
            const int m = lin >> 5, kk = lin & 31;
            const int brow = m0 + m;
            const float g0 = gate[brow * 3 + 0];
            const float g1 = gate[brow * 3 + 1];
            const float g2 = gate[brow * 3 + 2];
            const size_t off = (size_t)brow * H_DIM + k0 + kk;
            const float4 f0 = *(const float4*)(new_sub + 0 * SBH + off);
            const float4 f1 = *(const float4*)(new_sub + 1 * SBH + off);
            const float4 f2 = *(const float4*)(new_sub + 2 * SBH + off);
            As[m][kk + 0] = (_Float16)(g0 * f0.x + g1 * f1.x + g2 * f2.x);
            As[m][kk + 1] = (_Float16)(g0 * f0.y + g1 * f1.y + g2 * f2.y);
            As[m][kk + 2] = (_Float16)(g0 * f0.z + g1 * f1.z + g2 * f2.z);
            As[m][kk + 3] = (_Float16)(g0 * f0.w + g1 * f1.w + g2 * f2.w);
        }
        // stage B = fusion_W[n, k]
        #pragma unroll
        for (int r = 0; r < 2; ++r) {
            const int lin = tid * 4 + r * 1024;
            const int n = lin >> 5, kk = lin & 31;
            const float4 f = *(const float4*)(fusion_W + (size_t)(n0 + n) * H_DIM + k0 + kk);
            Bs[n][kk + 0] = (_Float16)f.x; Bs[n][kk + 1] = (_Float16)f.y;
            Bs[n][kk + 2] = (_Float16)f.z; Bs[n][kk + 3] = (_Float16)f.w;
        }
        __syncthreads();

        Frag16 a[2], b[2];
        #pragma unroll
        for (int mt = 0; mt < 2; ++mt) {
            const int row = wm + mt * 16 + mrow;
            #pragma unroll
            for (int j = 0; j < 8; ++j) {
                const int k = ((j < 4) ? 2 * j : 16 + 2 * (j - 4)) + half8;
                a[mt].u[j] = *(const unsigned int*)&As[row][k];
            }
        }
        #pragma unroll
        for (int nt = 0; nt < 2; ++nt) {
            const int col = wn + nt * 16 + mrow;
            #pragma unroll
            for (int j = 0; j < 8; ++j)
                b[nt].u[j] = *(const unsigned int*)&Bs[col][((j < 4) ? 2 * j : 16 + 2 * (j - 4)) + half8];
        }
        #pragma unroll
        for (int mt = 0; mt < 2; ++mt)
            #pragma unroll
            for (int nt = 0; nt < 2; ++nt)
                acc[mt][nt] = __builtin_amdgcn_wmma_f32_16x16x32_f16(
                    false, a[mt].v, false, b[nt].v,
                    (short)0, acc[mt][nt], false, false);
    }

    const int mhi = (lane >> 4) * 8;
    #pragma unroll
    for (int mt = 0; mt < 2; ++mt) {
        #pragma unroll
        for (int nt = 0; nt < 2; ++nt) {
            const int n = n0 + wn + nt * 16 + mrow;
            const float bias = fusion_b[n];
            #pragma unroll
            for (int j = 0; j < 8; ++j) {
                const int m = m0 + wm + mt * 16 + j + mhi;
                pre[(size_t)m * H_DIM + n] =
                    prev_master_h[(size_t)m * H_DIM + n] + acc[mt][nt][j] + bias;
            }
        }
    }
}

// ---------------------------------------------------------------------------
// Kernel 4: in-place LayerNorm + sigmoid over each row (each block owns a row;
// all reads of the row happen before any writes -> safe in place).
// ---------------------------------------------------------------------------
__global__ __launch_bounds__(256)
void ln_sigmoid_kernel(float* __restrict__ buf,        // [B,H] pre -> master_h
                       const float* __restrict__ ln_g,
                       const float* __restrict__ ln_b)
{
    __shared__ float red[256];
    const int b = blockIdx.x, tid = threadIdx.x;

    float vals[8];
    float s = 0.f;
    #pragma unroll
    for (int r = 0; r < 8; ++r) {
        vals[r] = buf[(size_t)b * H_DIM + tid + r * 256];
        s += vals[r];
    }
    red[tid] = s; __syncthreads();
    for (int off = 128; off > 0; off >>= 1) {
        if (tid < off) red[tid] += red[tid + off];
        __syncthreads();
    }
    const float mu = red[0] * (1.f / H_DIM);
    __syncthreads();

    float sq = 0.f;
    #pragma unroll
    for (int r = 0; r < 8; ++r) { const float d = vals[r] - mu; sq += d * d; }
    red[tid] = sq; __syncthreads();
    for (int off = 128; off > 0; off >>= 1) {
        if (tid < off) red[tid] += red[tid + off];
        __syncthreads();
    }
    const float rinv = rsqrtf(red[0] * (1.f / H_DIM) + LN_EPS);

    #pragma unroll
    for (int r = 0; r < 8; ++r) {
        const int h = tid + r * 256;
        const float nrm = (vals[r] - mu) * rinv * ln_g[h] + ln_b[h];
        buf[(size_t)b * H_DIM + h] = 1.f / (1.f + expf(-nrm));
    }
}

// ---------------------------------------------------------------------------
extern "C" void kernel_launch(void* const* d_in, const int* in_sizes, int n_in,
                              void* d_out, int out_size, void* d_ws, size_t ws_size,
                              hipStream_t stream)
{
    const float* x        = (const float*)d_in[0];
    const float* pmh      = (const float*)d_in[1];
    const float* psh      = (const float*)d_in[2];
    const float* gumbel   = (const float*)d_in[3];
    const float* W_ih     = (const float*)d_in[4];
    const float* W_hh     = (const float*)d_in[5];
    const float* b_ih     = (const float*)d_in[6];
    const float* b_hh     = (const float*)d_in[7];
    const float* gate_W   = (const float*)d_in[8];
    const float* gate_b   = (const float*)d_in[9];
    const float* fusion_W = (const float*)d_in[10];
    const float* fusion_b = (const float*)d_in[11];
    const float* ln_g     = (const float*)d_in[12];
    const float* ln_b     = (const float*)d_in[13];

    float* out_master = (float*)d_out;                               // [B,H]
    float* out_gate   = out_master + (size_t)B_DIM * H_DIM;          // [B,S]
    float* out_newsub = out_gate + (size_t)B_DIM * S_DIM;            // [S,B,H]

    gate_kernel<<<B_DIM, 256, 0, stream>>>(pmh, gate_W, gate_b, gumbel, out_gate);

    dim3 g1(B_DIM / BM, H_DIM / BN, S_DIM);
    sub_gemm_kernel<<<g1, 256, 0, stream>>>(x, psh, W_ih, W_hh, b_ih, b_hh, out_newsub);

    dim3 g2(B_DIM / BM, H_DIM / BN, 1);
    fusion_gemm_kernel<<<g2, 256, 0, stream>>>(out_newsub, out_gate, fusion_W,
                                               fusion_b, pmh, out_master);

    ln_sigmoid_kernel<<<B_DIM, 256, 0, stream>>>(out_master, ln_g, ln_b);
}
